// DynSMHALayer_19164144074772
// MI455X (gfx1250) — matrix operand order repaired and checked by reading, not compiled, not run on hardware
//
#include <hip/hip_runtime.h>
#include <hip/hip_bf16.h>
#include <math.h>

// Problem constants (from reference)
#define HIDDEN   1024
#define HEAD_DIM 64
#define NEXP     16
#define BATCH    4
#define TSEQ     2048
#define NTOK     (BATCH * TSEQ)   // 8192

typedef __bf16 bf16;
typedef __attribute__((ext_vector_type(16))) __bf16 v16bf;
typedef __attribute__((ext_vector_type(8)))  __bf16 v8bf;
typedef __attribute__((ext_vector_type(8)))  float  v8f;

union V16 { v16bf v; v8bf h[2]; };

#define WMMA_BF16(C, A, B) \
  (C) = __builtin_amdgcn_wmma_f32_16x16x32_bf16(false, (A), false, (B), (short)0, (C), false, false)

__device__ __forceinline__ bf16 f2bf(float f) { return (bf16)f; }  // RNE

// ---------------------------------------------------------------------------
// CDNA5 async global->LDS copies (ASYNCcnt), guarded so compile never breaks.
// Builtin signature (from hipcc diagnostic): param0 = v4i __device__(AS1)*,
// so we pass address-space-qualified v4i pointers.
// ---------------------------------------------------------------------------
#if defined(__gfx1250__) && \
    __has_builtin(__builtin_amdgcn_global_load_async_to_lds_b128) && \
    __has_builtin(__builtin_amdgcn_s_wait_asynccnt)
#define HAVE_ASYNC_LDS 1
#else
#define HAVE_ASYNC_LDS 0
#endif

typedef int v4i __attribute__((vector_size(4 * sizeof(int))));
typedef __attribute__((address_space(1))) v4i as1_v4i;
typedef __attribute__((address_space(3))) v4i as3_v4i;

// copy 16 bytes global -> LDS (async DMA when available)
__device__ __forceinline__ void cp_b128(void* lds, const void* g) {
#if HAVE_ASYNC_LDS
  __builtin_amdgcn_global_load_async_to_lds_b128((as1_v4i*)g, (as3_v4i*)lds, 0, 0);
#else
  *(uint4*)lds = *(const uint4*)g;
#endif
}

__device__ __forceinline__ void wait_async0() {
#if HAVE_ASYNC_LDS
  __builtin_amdgcn_s_wait_asynccnt(0);
#endif
}

// ---------------------------------------------------------------------------
// Kernel 0: per-expert sim-matrix column L2 norms + sigmoid(gates). 16 waves.
// ---------------------------------------------------------------------------
__global__ __launch_bounds__(512) void simstats_kernel(
    const float* __restrict__ sim, const float* __restrict__ gates,
    float* __restrict__ snorm, float* __restrict__ sgate) {
  int w = threadIdx.x >> 5, lane = threadIdx.x & 31;
  float s = 0.f;
  for (int c = lane; c < HIDDEN; c += 32) {
    float v = sim[c * NEXP + w];
    s += v * v;
  }
  #pragma unroll
  for (int off = 16; off > 0; off >>= 1) s += __shfl_xor(s, off, 32);
  if (lane == 0) {
    snorm[w] = fmaxf(sqrtf(s), 1e-12f);
    sgate[w] = 1.f / (1.f + __expf(-gates[w]));
  }
}

// ---------------------------------------------------------------------------
// Kernel 1a: q/k/v weights (e,k,d) -> bf16 fragment-native layout
//   dst[((e*32 + k/32)*64 + d)*32 + k%32]
// so a WMMA B-fragment (col d, K=kb*32+16h+i, i=0..15) is 16 contiguous bf16.
// ---------------------------------------------------------------------------
__global__ __launch_bounds__(256) void cvt_qkv_t(
    const float* __restrict__ src, bf16* __restrict__ dst) {
  int i = blockIdx.x * 256 + threadIdx.x;           // n = 16*1024*64
  int d = i & 63;
  int k = (i >> 6) & 1023;
  int e = i >> 16;
  dst[((e * 32 + (k >> 5)) * 64 + d) * 32 + (k & 31)] = f2bf(src[i]);
}

// Kernel 1b: o weights (e,d,c) -> dst[((e*2 + d/32)*1024 + c)*32 + d%32]
__global__ __launch_bounds__(256) void cvt_o_t(
    const float* __restrict__ src, bf16* __restrict__ dst) {
  int i = blockIdx.x * 256 + threadIdx.x;
  int c = i & 1023;
  int d = (i >> 10) & 63;
  int e = i >> 16;
  dst[((e * 2 + (d >> 5)) * 1024 + c) * 32 + (d & 31)] = f2bf(src[i]);
}

// ---------------------------------------------------------------------------
// Kernel 2: gating. One wave per token; shfl-reduced dots, lane 0 finishes.
// ---------------------------------------------------------------------------
__global__ __launch_bounds__(256) void gate_kernel(
    const float* __restrict__ X, const float* __restrict__ sim,
    const float* __restrict__ snorm, const float* __restrict__ sgate,
    float* __restrict__ rw) {
  int w = threadIdx.x >> 5, lane = threadIdx.x & 31;
  int t = blockIdx.x * 8 + w;
  float acc[NEXP];
  #pragma unroll
  for (int e = 0; e < NEXP; ++e) acc[e] = 0.f;
  float xs = 0.f;
  for (int c = lane; c < HIDDEN; c += 32) {
    float x = X[t * HIDDEN + c];
    xs += x * x;
    #pragma unroll
    for (int e = 0; e < NEXP; ++e) acc[e] += x * sim[c * NEXP + e];
  }
  #pragma unroll
  for (int off = 16; off > 0; off >>= 1) {
    xs += __shfl_xor(xs, off, 32);
    #pragma unroll
    for (int e = 0; e < NEXP; ++e) acc[e] += __shfl_xor(acc[e], off, 32);
  }
  if (lane == 0) {
    float nx = fmaxf(sqrtf(xs), 1e-12f);
    float logit[NEXP];
    int active[NEXP];
    int na = 0;
    #pragma unroll
    for (int e = 0; e < NEXP; ++e) {
      float l = acc[e] / (nx * snorm[e]) - sgate[e];
      logit[e] = l;
      active[e] = (l > 0.f) ? 1 : 0;
      na += active[e];
    }
    if (na == 0) {  // top-2 fallback (first index wins ties, like lax.top_k)
      int i1 = 0;
      for (int e = 1; e < NEXP; ++e) if (logit[e] > logit[i1]) i1 = e;
      int i2 = (i1 == 0) ? 1 : 0;
      for (int e = 0; e < NEXP; ++e)
        if (e != i1 && logit[e] > logit[i2]) i2 = e;
      active[i1] = 1; active[i2] = 1;
    }
    float m = -3.4e38f;
    #pragma unroll
    for (int e = 0; e < NEXP; ++e)
      if (active[e]) m = fmaxf(m, fmaxf(logit[e], 0.f));
    float s = 0.f, ex[NEXP];
    #pragma unroll
    for (int e = 0; e < NEXP; ++e) {
      ex[e] = active[e] ? __expf(fmaxf(logit[e], 0.f) - m) : 0.f;
      s += ex[e];
    }
    float inv = 1.f / s;
    #pragma unroll
    for (int e = 0; e < NEXP; ++e) rw[t * NEXP + e] = ex[e] * inv;
  }
}

// ---------------------------------------------------------------------------
// Kernel 3: fused QKV. Gate folded into A fragments; one accumulator set per
// output spans all 16 experts. 8 waves (4x2) -> 64x64 tile of q, k and v.
// X tiles double-buffered via async global->LDS DMA; B fragments are two
// global_load_b128 each from the fragment-native weight layout (L2-resident).
// ---------------------------------------------------------------------------
__global__ __launch_bounds__(256) void qkv_kernel(
    const float* __restrict__ X, const float* __restrict__ rw,
    const bf16* __restrict__ Wq, const bf16* __restrict__ Wk,
    const bf16* __restrict__ Wv,
    bf16* __restrict__ qo, bf16* __restrict__ ko, bf16* __restrict__ vo) {
  __shared__ float Xs[2][64][36];   // stride 144B: 16B-aligned rows
  __shared__ float rws[64][17];
  const int m0 = blockIdx.x * 64;
  const int tid = threadIdx.x;
  const int w = tid >> 5, lane = tid & 31;
  const int wm = (w & 3) * 16, wn = (w >> 2) * 32;
  const int h = lane >> 4, ln = lane & 15;
  const int row = wm + ln;

  auto stage = [&](int buf, int kb) {
    for (int idx = tid; idx < 64 * 8; idx += 256) {
      int r = idx >> 3, c4 = (idx & 7) * 4;
      cp_b128(&Xs[buf][r][c4], &X[(m0 + r) * HIDDEN + kb * 32 + c4]);
    }
  };

  for (int idx = tid; idx < 64 * NEXP; idx += 256)
    rws[idx >> 4][idx & 15] = rw[(m0 + (idx >> 4)) * NEXP + (idx & 15)];
  stage(0, 0);
  __syncthreads();
  float rs[NEXP];
  #pragma unroll
  for (int e = 0; e < NEXP; ++e) rs[e] = rws[row][e];

  v8f cq0 = {}, cq1 = {}, ck0 = {}, ck1 = {}, cv0 = {}, cv1 = {};
  for (int kb = 0; kb < HIDDEN / 32; ++kb) {
    const int cur = kb & 1;
    wait_async0();       // this wave's DMA into Xs[cur] complete
    __syncthreads();     // all waves' DMA visible; prev reads of Xs[cur^1] done
    if (kb + 1 < HIDDEN / 32) stage(cur ^ 1, kb + 1);  // overlap with compute
    const float* xr = &Xs[cur][row][0];
    for (int e = 0; e < NEXP; ++e) {
      const float s = rs[e];
      v16bf a;
      #pragma unroll
      for (int j = 0; j < 8; ++j) {
        a[j]     = f2bf(s * xr[16 * h + j]);
        a[j + 8] = f2bf(s * xr[16 * h + 8 + j]);
      }
      // fragment base: lane column (wn+ln), K chunk 16h; +64 v8bf = col+16
      const size_t fo = (size_t)((e * 32 + kb) * 64 + wn + ln) * 32 + 16 * h;
      const v8bf* pq = (const v8bf*)(Wq + fo);
      const v8bf* pk = (const v8bf*)(Wk + fo);
      const v8bf* pv = (const v8bf*)(Wv + fo);
      V16 bq0, bq1, bk0, bk1, bv0, bv1;
      bq0.h[0] = pq[0];  bq0.h[1] = pq[1];
      bq1.h[0] = pq[64]; bq1.h[1] = pq[65];
      bk0.h[0] = pk[0];  bk0.h[1] = pk[1];
      bk1.h[0] = pk[64]; bk1.h[1] = pk[65];
      bv0.h[0] = pv[0];  bv0.h[1] = pv[1];
      bv1.h[0] = pv[64]; bv1.h[1] = pv[65];
      WMMA_BF16(cq0, a, bq0.v); WMMA_BF16(cq1, a, bq1.v);
      WMMA_BF16(ck0, a, bk0.v); WMMA_BF16(ck1, a, bk1.v);
      WMMA_BF16(cv0, a, bv0.v); WMMA_BF16(cv1, a, bv1.v);
    }
  }
  #pragma unroll
  for (int i = 0; i < 8; ++i) {
    int g = (m0 + wm + 8 * h + i) * HEAD_DIM + wn + ln;
    qo[g] = f2bf(cq0[i]); qo[g + 16] = f2bf(cq1[i]);
    ko[g] = f2bf(ck0[i]); ko[g + 16] = f2bf(ck1[i]);
    vo[g] = f2bf(cv0[i]); vo[g + 16] = f2bf(cv1[i]);
  }
}

// ---------------------------------------------------------------------------
// Kernel 4: causal flash attention, 64-query blocks. Q/K staged by async DMA;
// V transposed in LDS so both QK^T and PV fragments are ds_load_b128 pairs.
// ---------------------------------------------------------------------------
__global__ __launch_bounds__(256) void attn_kernel(
    const bf16* __restrict__ qm, const bf16* __restrict__ km,
    const bf16* __restrict__ vm, float* __restrict__ attn) {
  __shared__ bf16  Qs[64][72];   // stride 144B, 16B-aligned rows
  __shared__ bf16  Ks[64][72];
  __shared__ bf16  Vts[64][72];  // V transposed: Vts[d][key]
  __shared__ float Ss[64][64];
  __shared__ bf16  Ps[64][72];
  __shared__ float mrow[64], lrow[64], arow[64];

  const int b = blockIdx.y, qb = blockIdx.x;
  const int tq0 = b * TSEQ + qb * 64;
  const int tid = threadIdx.x;
  const int w = tid >> 5, lane = tid & 31;
  const int wm = (w & 3) * 16, wn = (w >> 2) * 32;
  const int h = lane >> 4, ln = lane & 15;
  const float scale = 0.125f;  // 1/sqrt(64)

  for (int idx = tid; idx < 64 * 8; idx += 256) {
    int r = idx >> 3, c8 = (idx & 7) * 8;
    cp_b128(&Qs[r][c8], &qm[(tq0 + r) * HEAD_DIM + c8]);
  }
  if (tid < 64) { mrow[tid] = -3.4e38f; lrow[tid] = 0.f; }

  v8f o0 = {}, o1 = {};
  for (int kb = 0; kb <= qb; ++kb) {
    const int tk0 = b * TSEQ + kb * 64;
    __syncthreads();   // previous iteration's LDS reads complete
    for (int idx = tid; idx < 64 * 8; idx += 256) {
      int r = idx >> 3, c8 = (idx & 7) * 8;
      cp_b128(&Ks[r][c8], &km[(tk0 + r) * HEAD_DIM + c8]);
    }
    for (int idx = tid; idx < 64 * 64; idx += 256) {
      int r = idx >> 6, d = idx & 63;
      Vts[d][r] = vm[(tk0 + r) * HEAD_DIM + d];
    }
    wait_async0();
    __syncthreads();
    // S = Q K^T over D=64 (two chained 32-wide WMMA steps)
    v8f s0 = {}, s1 = {};
    #pragma unroll
    for (int c0 = 0; c0 < 64; c0 += 32) {
      V16 a, b0, b1;
      a.h[0]  = *(const v8bf*)&Qs[wm + ln][c0 + 16 * h];
      a.h[1]  = *(const v8bf*)&Qs[wm + ln][c0 + 16 * h + 8];
      b0.h[0] = *(const v8bf*)&Ks[wn + ln][c0 + 16 * h];
      b0.h[1] = *(const v8bf*)&Ks[wn + ln][c0 + 16 * h + 8];
      b1.h[0] = *(const v8bf*)&Ks[wn + 16 + ln][c0 + 16 * h];
      b1.h[1] = *(const v8bf*)&Ks[wn + 16 + ln][c0 + 16 * h + 8];
      WMMA_BF16(s0, a.v, b0.v); WMMA_BF16(s1, a.v, b1.v);
    }
    #pragma unroll
    for (int i = 0; i < 8; ++i) {
      Ss[wm + 8 * h + i][wn + ln]      = s0[i];
      Ss[wm + 8 * h + i][wn + 16 + ln] = s1[i];
    }
    __syncthreads();
    // online softmax bookkeeping (one thread per query row)
    if (tid < 64) {
      const int r = tid, qg = qb * 64 + r;
      float mnew = mrow[r];
      for (int j = 0; j < 64; ++j)
        if (kb * 64 + j <= qg) mnew = fmaxf(mnew, Ss[r][j] * scale);
      float al = __expf(mrow[r] - mnew), ls = 0.f;
      for (int j = 0; j < 64; ++j) {
        float p = (kb * 64 + j <= qg) ? __expf(Ss[r][j] * scale - mnew) : 0.f;
        Ps[r][j] = f2bf(p);
        ls += p;
      }
      lrow[r] = lrow[r] * al + ls;
      mrow[r] = mnew;
      arow[r] = al;
    }
    __syncthreads();
    // O = O*alpha + P V
    #pragma unroll
    for (int i = 0; i < 8; ++i) {
      float al = arow[wm + 8 * h + i];
      o0[i] *= al; o1[i] *= al;
    }
    #pragma unroll
    for (int c0 = 0; c0 < 64; c0 += 32) {
      V16 a, b0, b1;
      a.h[0]  = *(const v8bf*)&Ps[wm + ln][c0 + 16 * h];
      a.h[1]  = *(const v8bf*)&Ps[wm + ln][c0 + 16 * h + 8];
      b0.h[0] = *(const v8bf*)&Vts[wn + ln][c0 + 16 * h];
      b0.h[1] = *(const v8bf*)&Vts[wn + ln][c0 + 16 * h + 8];
      b1.h[0] = *(const v8bf*)&Vts[wn + 16 + ln][c0 + 16 * h];
      b1.h[1] = *(const v8bf*)&Vts[wn + 16 + ln][c0 + 16 * h + 8];
      WMMA_BF16(o0, a.v, b0.v); WMMA_BF16(o1, a.v, b1.v);
    }
  }
  #pragma unroll
  for (int i = 0; i < 8; ++i) {
    float inv = 1.f / lrow[wm + 8 * h + i];
    int g = (tq0 + wm + 8 * h + i) * HEAD_DIM + wn + ln;
    attn[g] = o0[i] * inv;
    attn[g + 16] = o1[i] * inv;
  }
}

// ---------------------------------------------------------------------------
// Kernel 5: output projection. out[m,:] = sum_e (rw_e[m]*attn[m,:]) @ Wo_e.
// ---------------------------------------------------------------------------
__global__ __launch_bounds__(256) void oproj_kernel(
    const float* __restrict__ attn, const float* __restrict__ rw,
    const bf16* __restrict__ Wo, float* __restrict__ out) {
  __shared__ float As[64][68];   // stride 272B, 16B-aligned rows
  __shared__ float rws[64][17];
  const int m0 = blockIdx.x * 64;
  const int n0 = blockIdx.y * 64;
  const int tid = threadIdx.x;
  const int w = tid >> 5, lane = tid & 31;
  const int wm = (w & 3) * 16, wn = (w >> 2) * 32;
  const int h = lane >> 4, ln = lane & 15;
  const int row = wm + ln;

  for (int idx = tid; idx < 64 * 16; idx += 256) {
    int r = idx >> 4, c4 = (idx & 15) * 4;
    cp_b128(&As[r][c4], &attn[(m0 + r) * HEAD_DIM + c4]);
  }
  for (int idx = tid; idx < 64 * NEXP; idx += 256)
    rws[idx >> 4][idx & 15] = rw[(m0 + (idx >> 4)) * NEXP + (idx & 15)];
  wait_async0();
  __syncthreads();
  float rs[NEXP];
  #pragma unroll
  for (int e = 0; e < NEXP; ++e) rs[e] = rws[row][e];
  const float* ar = &As[row][0];

  v8f c0 = {}, c1 = {};
  for (int e = 0; e < NEXP; ++e) {
    const float s = rs[e];
    #pragma unroll
    for (int kb2 = 0; kb2 < 2; ++kb2) {
      v16bf a;
      #pragma unroll
      for (int j = 0; j < 8; ++j) {
        a[j]     = f2bf(s * ar[kb2 * 32 + 16 * h + j]);
        a[j + 8] = f2bf(s * ar[kb2 * 32 + 16 * h + 8 + j]);
      }
      const size_t fo = (size_t)((e * 2 + kb2) * 1024 + n0 + wn + ln) * 32 + 16 * h;
      const v8bf* po = (const v8bf*)(Wo + fo);
      V16 b0, b1;
      b0.h[0] = po[0];  b0.h[1] = po[1];
      b1.h[0] = po[64]; b1.h[1] = po[65];
      WMMA_BF16(c0, a, b0.v); WMMA_BF16(c1, a, b1.v);
    }
  }
  #pragma unroll
  for (int i = 0; i < 8; ++i) {
    int g = (m0 + wm + 8 * h + i) * HIDDEN + n0 + wn + ln;
    out[g] = c0[i];
    out[g + 16] = c1[i];
  }
}

// ---------------------------------------------------------------------------
extern "C" void kernel_launch(void* const* d_in, const int* in_sizes, int n_in,
                              void* d_out, int out_size, void* d_ws, size_t ws_size,
                              hipStream_t stream) {
  const float* X     = (const float*)d_in[0];  // (4,2048,1024)
  const float* sim   = (const float*)d_in[1];  // (1024,16)
  const float* gates = (const float*)d_in[2];  // (16,)
  const float* q_w   = (const float*)d_in[3];  // (16,1024,64)
  const float* k_w   = (const float*)d_in[4];
  const float* v_w   = (const float*)d_in[5];
  const float* o_w   = (const float*)d_in[6];  // (16,64,1024)
  float* out = (float*)d_out;

  char* ws = (char*)d_ws;
  size_t off = 0;
  auto take = [&](size_t bytes) {
    void* p = ws + off;
    off = (off + bytes + 255) & ~(size_t)255;
    return p;
  };
  float* rw    = (float*)take((size_t)NTOK * NEXP * 4);
  float* snorm = (float*)take(16 * 4);
  float* sgate = (float*)take(16 * 4);
  bf16* Wq = (bf16*)take((size_t)NEXP * HIDDEN * HEAD_DIM * 2);
  bf16* Wk = (bf16*)take((size_t)NEXP * HIDDEN * HEAD_DIM * 2);
  bf16* Wv = (bf16*)take((size_t)NEXP * HIDDEN * HEAD_DIM * 2);
  bf16* Wo = (bf16*)take((size_t)NEXP * HEAD_DIM * HIDDEN * 2);
  bf16* qb = (bf16*)take((size_t)NTOK * HEAD_DIM * 2);
  bf16* kb = (bf16*)take((size_t)NTOK * HEAD_DIM * 2);
  bf16* vb = (bf16*)take((size_t)NTOK * HEAD_DIM * 2);
  float* at = (float*)take((size_t)NTOK * HEAD_DIM * 4);
  (void)ws_size; (void)in_sizes; (void)n_in; (void)out_size;

  const int wn = NEXP * HIDDEN * HEAD_DIM;  // 1048576 elems per weight tensor
  simstats_kernel<<<1, 512, 0, stream>>>(sim, gates, snorm, sgate);
  cvt_qkv_t<<<wn / 256, 256, 0, stream>>>(q_w, Wq);
  cvt_qkv_t<<<wn / 256, 256, 0, stream>>>(k_w, Wk);
  cvt_qkv_t<<<wn / 256, 256, 0, stream>>>(v_w, Wv);
  cvt_o_t  <<<wn / 256, 256, 0, stream>>>(o_w, Wo);
  gate_kernel<<<NTOK / 8, 256, 0, stream>>>(X, sim, snorm, sgate, rw);
  qkv_kernel<<<NTOK / 64, 256, 0, stream>>>(X, rw, Wq, Wk, Wv, qb, kb, vb);
  attn_kernel<<<dim3(TSEQ / 64, BATCH), 256, 0, stream>>>(qb, kb, vb, at);
  oproj_kernel<<<dim3(NTOK / 64, HIDDEN / 64), 256, 0, stream>>>(at, rw, Wo, out);
}
